// SelectiveMemoryUnit_57028575756937
// MI455X (gfx1250) — matrix operand order repaired and checked by reading, not compiled, not run on hardware
//
#include <hip/hip_runtime.h>

#define B_ 8
#define T_ 2048
#define D_ 512
#define S_ 16
#define NEGF (-1.0e30f)

typedef unsigned short u16;
typedef __attribute__((ext_vector_type(16))) __bf16 v16bf;
typedef __attribute__((ext_vector_type(8)))  float  v8f;
typedef __attribute__((ext_vector_type(4)))  unsigned int u32x4;
typedef __attribute__((ext_vector_type(2)))  unsigned int u32x2;

union FragU { u32x4 u[2]; v16bf v; };

__device__ __forceinline__ int lane_id() { return (int)(threadIdx.x & 31u); }

__device__ __forceinline__ u16 f2bf(float f) {
  __bf16 h = (__bf16)f;
  return __builtin_bit_cast(u16, h);
}

// A fragment (16x32 bf16, M x K). lane = kb*16 + m.
// VGPR0-3: K = kb*8 .. kb*8+7 ; VGPR4-7: K = 16+kb*8 .. 16+kb*8+7.
__device__ __forceinline__ v16bf frag_a(const u16* base, int ld) {
  int l = lane_id(); int m = l & 15; int kb = l >> 4;
  FragU f;
  const u16* p = base + m * ld + kb * 8;
  f.u[0] = *(const u32x4*)(p);
  f.u[1] = *(const u32x4*)(p + 16);
  return f.v;
}

// B fragment (32x16 bf16, K x N) where B[k][n] = W[n][k] (W row-major).
// lane = kh*16 + n; VGPR0-3: K = kh*8 .. kh*8+7 ; VGPR4-7: K = 16+kh*8 .. 16+kh*8+7.
__device__ __forceinline__ v16bf frag_b(const u16* base, int ld) {
  int l = lane_id(); int n = l & 15; int kh = l >> 4;
  FragU f;
  const u16* p = base + n * ld + kh * 8;
  f.u[0] = *(const u32x4*)(p);
  f.u[1] = *(const u32x4*)(p + 16);
  return f.v;
}

// A fragment from 8 real rows in LDS (rows 8..15 forced to zero).
__device__ __forceinline__ v16bf frag_a_h(const u16* hbf, int k) {
  int l = lane_id(); int m = l & 15; int kb = l >> 4;
  FragU f;
  u32x4 z = {};
  if (m < 8) {
    const u16* p = hbf + m * D_ + k + kb * 8;
    f.u[0] = *(const u32x4*)(p);
    f.u[1] = *(const u32x4*)(p + 16);
  } else {
    f.u[0] = z; f.u[1] = z;
  }
  return f.v;
}

__device__ __forceinline__ v8f wmma_bf16(v16bf a, v16bf b, v8f c) {
  return __builtin_amdgcn_wmma_f32_16x16x32_bf16(false, a, false, b, (short)0, c,
                                                 false, false);
}

// ---------------------------------------------------------------- K0: convert
__global__ void f32_to_bf16_kern(const float* __restrict__ in,
                                 u16* __restrict__ out, int n) {
  int i = (int)(blockIdx.x * 256 + threadIdx.x);
  if (i < n) out[i] = f2bf(in[i]);
}

// --------------------------------------------------- K1/K3: out = X @ W^T + b
// X: [M,K] bf16 row-major; W: [N,K] bf16 row-major; 1 wave/block, 16x64 tile.
template <bool BF16OUT>
__global__ __launch_bounds__(32) void gemm_xwT(const u16* __restrict__ X,
                                               const u16* __restrict__ W,
                                               const float* __restrict__ bias,
                                               void* __restrict__ outp,
                                               int N, int K) {
  int mt = (int)blockIdx.x * 16;
  int nt = (int)blockIdx.y * 64;
  v8f acc[4] = {};
  for (int k = 0; k < K; k += 32) {
    v16bf a = frag_a(X + (size_t)mt * K + k, K);
#pragma unroll
    for (int i = 0; i < 4; ++i) {
      v16bf b = frag_b(W + (size_t)(nt + 16 * i) * K + k, K);
      acc[i] = wmma_bf16(a, b, acc[i]);
    }
  }
  int l = lane_id(); int n = l & 15; int half = l >> 4;
#pragma unroll
  for (int i = 0; i < 4; ++i) {
    int col = nt + 16 * i + n;
    float bv = bias[col];
#pragma unroll
    for (int r = 0; r < 8; ++r) {
      int row = mt + half * 8 + r;
      float v = acc[i][r] + bv;
      if (BF16OUT) ((u16*)outp)[(size_t)row * N + col] = f2bf(v);
      else         ((float*)outp)[(size_t)row * N + col] = v;
    }
  }
}

// ------------------------------------- K2: fused scores + top-5 + summary
// grid = (T/16, B), block = 128 (4 waves). Wave w handles j-tiles w, w+4, ...
__global__ __launch_bounds__(128) void score_topk_summary(
    const u16* __restrict__ qs, const u16* __restrict__ qc,
    const u16* __restrict__ ps, const u16* __restrict__ pc,
    const float* __restrict__ edu, u16* __restrict__ summary_bf) {
  __shared__ __align__(16) u16 qsL[16 * D_];
  __shared__ __align__(16) u16 qcL[16 * D_];
  __shared__ float tileL[4][16][16];
  __shared__ float mvL[4][16][5];
  __shared__ int   miL[4][16][5];
  __shared__ float fvL[16][5];
  __shared__ int   fiL[16][5];
  __shared__ float fcnt[16];

  int tid  = (int)threadIdx.x;
  int wave = tid >> 5, lane = tid & 31;
  int b  = (int)blockIdx.y;
  int tt = (int)blockIdx.x;
  int t0 = tt * 16;

  const u16* qsB = qs + ((size_t)(b * T_ + t0)) * D_;
  const u16* qcB = qc + ((size_t)(b * T_ + t0)) * D_;
  for (int i = tid; i < 16 * D_ / 4; i += 128) {
    ((u32x2*)qsL)[i] = ((const u32x2*)qsB)[i];
    ((u32x2*)qcL)[i] = ((const u32x2*)qcB)[i];
  }
  __syncthreads();

  float tv[5]; int tj[5];
#pragma unroll
  for (int s = 0; s < 5; ++s) { tv[s] = NEGF; tj[s] = 0; }

  for (int jt = wave; jt <= tt; jt += 4) {
    int j0 = jt * 16;
    const u16* psB = ps + ((size_t)(b * T_ + j0)) * D_;
    const u16* pcB = pc + ((size_t)(b * T_ + j0)) * D_;
    __builtin_prefetch(psB + 4 * 16 * D_, 0, 1);
    __builtin_prefetch(pcB + 4 * 16 * D_, 0, 1);
    v8f acc = {};
    for (int k = 0; k < D_; k += 32) {
      acc = wmma_bf16(frag_a(qsL + k, D_), frag_b(psB + k, D_), acc);
      acc = wmma_bf16(frag_a(qcL + k, D_), frag_b(pcB + k, D_), acc);
    }
    int n = lane & 15, half = lane >> 4;
#pragma unroll
    for (int r = 0; r < 8; ++r) {
      int m = half * 8 + r;
      float v = acc[r];
      if (j0 + n >= t0 + m) v = NEGF;  // strict causal: keep only j < t
      tileL[wave][m][n] = v;
    }
    asm volatile("s_wait_dscnt 0x0" ::: "memory");
    if (lane < 16) {
      int m = lane;
#pragma unroll
      for (int c = 0; c < 16; ++c) {
        float v = tileL[wave][m][c];
        if (v > tv[0]) {                       // sorted (ascending) insert
          tv[0] = v; tj[0] = j0 + c;
#pragma unroll
          for (int s = 0; s < 4; ++s) {
            if (tv[s] > tv[s + 1]) {
              float ftmp = tv[s]; tv[s] = tv[s + 1]; tv[s + 1] = ftmp;
              int   itmp = tj[s]; tj[s] = tj[s + 1]; tj[s + 1] = itmp;
            }
          }
        }
      }
    }
  }

  if (lane < 16) {
#pragma unroll
    for (int s = 0; s < 5; ++s) { mvL[wave][lane][s] = tv[s]; miL[wave][lane][s] = tj[s]; }
  }
  __syncthreads();

  if (tid < 16) {  // merge 4 waves' top-5 -> final top-5 per row
    int m = tid;
    float bv[5]; int bj[5];
#pragma unroll
    for (int s = 0; s < 5; ++s) { bv[s] = NEGF; bj[s] = 0; }
    for (int w = 0; w < 4; ++w) {
#pragma unroll
      for (int s = 0; s < 5; ++s) {
        float v = mvL[w][m][s]; int j = miL[w][m][s];
        if (v > bv[0]) {
          bv[0] = v; bj[0] = j;
#pragma unroll
          for (int s2 = 0; s2 < 4; ++s2) {
            if (bv[s2] > bv[s2 + 1]) {
              float ftmp = bv[s2]; bv[s2] = bv[s2 + 1]; bv[s2 + 1] = ftmp;
              int   itmp = bj[s2]; bj[s2] = bj[s2 + 1]; bj[s2 + 1] = itmp;
            }
          }
        }
      }
    }
    float cnt = 0.f;
#pragma unroll
    for (int s = 0; s < 5; ++s) {
      fvL[m][s] = bv[s]; fiL[m][s] = bj[s];
      if (bv[s] > 0.5f * NEGF) cnt += 1.f;
    }
    fcnt[m] = fmaxf(cnt, 1.f);
  }
  __syncthreads();

  // summary[b, t0+m, :] = mean of selected edu rows
  for (int e = tid; e < 16 * D_; e += 128) {
    int m = e >> 9, d = e & (D_ - 1);
    float sum = 0.f;
#pragma unroll
    for (int s = 0; s < 5; ++s) {
      if (fvL[m][s] > 0.5f * NEGF)
        sum += edu[((size_t)(b * T_ + fiL[m][s])) * D_ + d];
    }
    summary_bf[((size_t)(b * T_ + t0 + m)) * D_ + d] = f2bf(sum / fcnt[m]);
  }
}

// ------------------------------------------------ K4: sequential GRU scan
// Single block, 256 threads (8 waves). mem = d_out [B,S,D] f32, kept in L2.
__global__ __launch_bounds__(256) void scan_gru(
    const int* __restrict__ spk_all, const float* __restrict__ gi_all,
    const u16* __restrict__ Whh, const float* __restrict__ bhh,
    float* __restrict__ mem) {
  __shared__ __align__(16) u16 hbf[8 * D_];     // current h (bf16), rows 8..15 implicit 0
  __shared__ float ghL[8 * 3 * D_];             // gh for this step
  __shared__ int spkL[8];
  int tid = (int)threadIdx.x, wave = tid >> 5, lane = tid & 31;

  for (int i = tid; i < B_ * S_ * D_; i += 256) mem[i] = 0.f;
  __syncthreads();

  for (int t = 1; t < T_; ++t) {   // t==0 has no past -> no update
    if (tid < 8) spkL[tid] = spk_all[tid * T_ + t];
    __syncthreads();
    for (int i = tid; i < 8 * D_; i += 256) {
      int bb = i >> 9, d = i & (D_ - 1);
      hbf[i] = f2bf(mem[((size_t)bb * S_ + spkL[bb]) * D_ + d]);
    }
    __syncthreads();

    // gh = h @ W_hh^T + b_hh : [16(pad8), 512] x [512, 1536]
    for (int ntile = wave; ntile < 96; ntile += 8) {
      int n0 = ntile * 16;
      v8f acc = {};
      for (int k = 0; k < D_; k += 32)
        acc = wmma_bf16(frag_a_h(hbf, k), frag_b(Whh + (size_t)n0 * D_ + k, D_), acc);
      int n = lane & 15, half = lane >> 4;
      if (half == 0) {   // rows 0..7 are the real batch rows
        float bv = bhh[n0 + n];
#pragma unroll
        for (int r = 0; r < 8; ++r) ghL[r * (3 * D_) + n0 + n] = acc[r] + bv;
      }
    }
    __syncthreads();

    // gates
    for (int e = tid; e < 8 * D_; e += 256) {
      int bb = e >> 9, d = e & (D_ - 1);
      const float* gi = gi_all + ((size_t)(bb * T_ + t)) * (3 * D_);
      float ir = gi[d], iz = gi[D_ + d], inn = gi[2 * D_ + d];
      float hr = ghL[bb * (3 * D_) + d];
      float hz = ghL[bb * (3 * D_) + D_ + d];
      float hn = ghL[bb * (3 * D_) + 2 * D_ + d];
      float r = 1.f / (1.f + __expf(-(ir + hr)));
      float z = 1.f / (1.f + __expf(-(iz + hz)));
      float nn = tanhf(inn + r * hn);
      float* hp = &mem[((size_t)bb * S_ + spkL[bb]) * D_ + d];
      float h = *hp;
      *hp = (1.f - z) * nn + z * h;
    }
    __syncthreads();
  }
}

// ------------------------------------------------------------------ launcher
extern "C" void kernel_launch(void* const* d_in, const int* in_sizes, int n_in,
                              void* d_out, int out_size, void* d_ws, size_t ws_size,
                              hipStream_t stream) {
  (void)in_sizes; (void)n_in; (void)out_size; (void)ws_size;
  const float* edu  = (const float*)d_in[0];
  const int*   spk  = (const int*)d_in[1];
  const float* Wq_s = (const float*)d_in[2];
  const float* bq_s = (const float*)d_in[3];
  const float* Wq_c = (const float*)d_in[4];
  const float* bq_c = (const float*)d_in[5];
  const float* Wp_s = (const float*)d_in[6];
  const float* bp_s = (const float*)d_in[7];
  const float* Wp_c = (const float*)d_in[8];
  const float* bp_c = (const float*)d_in[9];
  const float* W_ih = (const float*)d_in[10];
  const float* W_hh = (const float*)d_in[11];
  const float* b_ih = (const float*)d_in[12];
  const float* b_hh = (const float*)d_in[13];

  char* ws = (char*)d_ws;
  size_t off = 0;
  auto alloc = [&](size_t bytes) -> void* {
    void* p = ws + off;
    off = (off + bytes + 255) & ~(size_t)255;
    return p;
  };
  const size_t MT = (size_t)B_ * T_;          // 16384 rows
  u16* edu_bf = (u16*)alloc(MT * D_ * 2);
  u16* qs_bf  = (u16*)alloc(MT * D_ * 2);
  u16* qc_bf  = (u16*)alloc(MT * D_ * 2);
  u16* ps_bf  = (u16*)alloc(MT * D_ * 2);
  u16* pc_bf  = (u16*)alloc(MT * D_ * 2);
  u16* Wqs_bf = (u16*)alloc((size_t)D_ * D_ * 2);
  u16* Wqc_bf = (u16*)alloc((size_t)D_ * D_ * 2);
  u16* Wps_bf = (u16*)alloc((size_t)D_ * D_ * 2);
  u16* Wpc_bf = (u16*)alloc((size_t)D_ * D_ * 2);
  u16* Wih_bf = (u16*)alloc((size_t)3 * D_ * D_ * 2);
  u16* Whh_bf = (u16*)alloc((size_t)3 * D_ * D_ * 2);
  u16* sum_bf = (u16*)alloc(MT * D_ * 2);
  float* gi   = (float*)alloc(MT * (size_t)(3 * D_) * 4);

  auto cvt = [&](const float* src, u16* dst, size_t n) {
    f32_to_bf16_kern<<<dim3((unsigned)((n + 255) / 256)), dim3(256), 0, stream>>>(
        src, dst, (int)n);
  };
  cvt(edu, edu_bf, MT * D_);
  cvt(Wq_s, Wqs_bf, (size_t)D_ * D_);
  cvt(Wq_c, Wqc_bf, (size_t)D_ * D_);
  cvt(Wp_s, Wps_bf, (size_t)D_ * D_);
  cvt(Wp_c, Wpc_bf, (size_t)D_ * D_);
  cvt(W_ih, Wih_bf, (size_t)3 * D_ * D_);
  cvt(W_hh, Whh_bf, (size_t)3 * D_ * D_);

  // K1: four projections (bf16 out)
  dim3 g1((unsigned)(MT / 16), (unsigned)(D_ / 64));
  gemm_xwT<true><<<g1, 32, 0, stream>>>(edu_bf, Wqs_bf, bq_s, qs_bf, D_, D_);
  gemm_xwT<true><<<g1, 32, 0, stream>>>(edu_bf, Wqc_bf, bq_c, qc_bf, D_, D_);
  gemm_xwT<true><<<g1, 32, 0, stream>>>(edu_bf, Wps_bf, bp_s, ps_bf, D_, D_);
  gemm_xwT<true><<<g1, 32, 0, stream>>>(edu_bf, Wpc_bf, bp_c, pc_bf, D_, D_);

  // K2: fused scores + top-5 + summary
  dim3 g2((unsigned)(T_ / 16), (unsigned)B_);
  score_topk_summary<<<g2, 128, 0, stream>>>(qs_bf, qc_bf, ps_bf, pc_bf, edu, sum_bf);

  // K3: gi = summary @ W_ih^T + b_ih (f32 out)
  dim3 g3((unsigned)(MT / 16), (unsigned)((3 * D_) / 64));
  gemm_xwT<false><<<g3, 32, 0, stream>>>(sum_bf, Wih_bf, b_ih, gi, 3 * D_, D_);

  // K4: sequential scan -> d_out
  scan_gru<<<1, 256, 0, stream>>>(spk, gi, Whh_bf, b_hh, (float*)d_out);
}
